// MultiHeadSelfAttention_63144609186407
// MI455X (gfx1250) — matrix operand order repaired and checked by reading, not compiled, not run on hardware
//
#include <hip/hip_runtime.h>
#include <hip/hip_bf16.h>

typedef _Float16 half_t;
typedef __attribute__((ext_vector_type(16))) _Float16 v16h;
typedef __attribute__((ext_vector_type(4)))  _Float16 v4h;
typedef __attribute__((ext_vector_type(8)))  float    v8f;
typedef __attribute__((ext_vector_type(4)))  int      v4i_t;

#define WMMA_F16(a, b, c) \
  __builtin_amdgcn_wmma_f32_16x16x32_f16(false, (a), false, (b), (short)0, (c), false, false)

// ---- CDNA5 async global->LDS copy (ASYNCcnt path), guarded ----
#if defined(__has_builtin)
#if __has_builtin(__builtin_amdgcn_global_load_async_to_lds_b128)
#define HAVE_ASYNC_LDS 1
#endif
#endif

#ifdef HAVE_ASYNC_LDS
#if __has_builtin(__builtin_amdgcn_s_wait_asynccnt)
#define ASYNC_WAIT() __builtin_amdgcn_s_wait_asynccnt(0)
#else
#define ASYNC_WAIT() asm volatile("s_wait_asynccnt 0x0" ::: "memory")
#endif
#endif

static constexpr int B_  = 2;
static constexpr int S_  = 2048;
static constexpr int D_  = 1024;
static constexpr int H_  = 16;
static constexpr int DH_ = 64;

__device__ __forceinline__ v8f v8f_zero() {
  v8f z = {0.f, 0.f, 0.f, 0.f, 0.f, 0.f, 0.f, 0.f};
  return z;
}

// A fragment 16x32 f16 from row-major src (p = &src[row0][col0], leading dim ld).
// Lane l holds row M = l&15; khalf = (l>>4)*16; element e: k = khalf + e.
__device__ __forceinline__ v16h load_a_frag(const half_t* __restrict__ p, int ld) {
  int lane = threadIdx.x & 31;
  const half_t* r = p + (lane & 15) * ld + ((lane >> 4) << 4);
  v16h a;
#pragma unroll
  for (int e = 0; e < 16; ++e) a[e] = r[e];
  return a;
}

// B fragment 32x16 where B[k][n] = srcT[n][k]; srcT row-major, p = &srcT[n0][k0].
// Lane l: n = l&15, khalf = (l>>4)*16; element e: k = khalf + e.  (contiguous 32B/lane)
__device__ __forceinline__ v16h load_bT_frag(const half_t* __restrict__ p, int ld) {
  int lane = threadIdx.x & 31;
  const half_t* r = p + (lane & 15) * ld + ((lane >> 4) << 4);
  v16h b;
#pragma unroll
  for (int e = 0; e < 16; ++e) b[e] = r[e];
  return b;
}

// ---------------- fp32 -> fp16 convert (x4 vectorized) ----------------
__global__ void cvt_f32_f16_kernel(const float* __restrict__ in,
                                   half_t* __restrict__ out, int n4) {
  int i = blockIdx.x * blockDim.x + threadIdx.x;
  if (i < n4) {
    float4 v = ((const float4*)in)[i];
    v4h h;
    h[0] = (half_t)v.x; h[1] = (half_t)v.y; h[2] = (half_t)v.z; h[3] = (half_t)v.w;
    ((v4h*)out)[i] = h;
  }
}

// ---------------- QKV projection: out = X @ W^T + b ----------------
// Block: 8 waves share one 32-row X tile (staged to LDS via async copy);
// wave w handles head h = (blockIdx&1)*8 + w. 256 blocks.
// VT_OUT=false: out[b][h][s][dh] ; VT_OUT=true: out[b][h][dh][s] (for V).
template <bool VT_OUT>
__global__ __launch_bounds__(256)
void qkv_proj_kernel(const half_t* __restrict__ X, const half_t* __restrict__ W,
                     const float* __restrict__ bias, half_t* __restrict__ out) {
  __shared__ __align__(16) half_t xtile[32 * D_];  // 64 KB
  int wave = threadIdx.x >> 5, lane = threadIdx.x & 31;
  int mtile = blockIdx.x >> 1;
  int h = ((blockIdx.x & 1) << 3) + wave;
  int mg0 = mtile * 32;
  int b = mg0 >> 11;
  int s0 = mg0 & (S_ - 1);

  // stage shared X tile (32 x 1024 f16 = 64 KB) into LDS
  const char* gsrc = (const char*)(X + (size_t)mg0 * D_);
#ifdef HAVE_ASYNC_LDS
  for (int c = threadIdx.x; c < 4096; c += 256)
    __builtin_amdgcn_global_load_async_to_lds_b128(
        (v4i_t*)(gsrc + (size_t)c * 16),
        (v4i_t*)((char*)xtile + c * 16), 0, 0);
  ASYNC_WAIT();
#else
  for (int c = threadIdx.x; c < 4096; c += 256)
    ((float4*)xtile)[c] = ((const float4*)gsrc)[c];
#endif
  __syncthreads();

  v8f acc[2][4];
#pragma unroll
  for (int a = 0; a < 2; ++a)
#pragma unroll
    for (int t = 0; t < 4; ++t) acc[a][t] = v8f_zero();

  const half_t* wbase = W + (size_t)h * DH_ * D_;

  // software pipeline: A frags from LDS, B frags double-buffered from global
  v16h a_cur[2], b_cur[4];
  a_cur[0] = load_a_frag(xtile, D_);
  a_cur[1] = load_a_frag(xtile + 16 * D_, D_);
#pragma unroll
  for (int t = 0; t < 4; ++t)
    b_cur[t] = load_bT_frag(wbase + (size_t)t * 16 * D_, D_);

  for (int k0 = 0; k0 < D_; k0 += 32) {
    int k1 = (k0 + 32) & (D_ - 1);  // wraps on last iter (harmless reload)
    v16h a_nxt[2], b_nxt[4];
    a_nxt[0] = load_a_frag(xtile + k1, D_);
    a_nxt[1] = load_a_frag(xtile + 16 * D_ + k1, D_);
#pragma unroll
    for (int t = 0; t < 4; ++t)
      b_nxt[t] = load_bT_frag(wbase + (size_t)t * 16 * D_ + k1, D_);

#pragma unroll
    for (int t = 0; t < 4; ++t) {
      acc[0][t] = WMMA_F16(a_cur[0], b_cur[t], acc[0][t]);
      acc[1][t] = WMMA_F16(a_cur[1], b_cur[t], acc[1][t]);
    }
    a_cur[0] = a_nxt[0]; a_cur[1] = a_nxt[1];
#pragma unroll
    for (int t = 0; t < 4; ++t) b_cur[t] = b_nxt[t];
  }

  int hf = lane >> 4, nn = lane & 15;
#pragma unroll
  for (int t = 0; t < 4; ++t) {
    int dh = t * 16 + nn;
    float bb = bias[h * DH_ + dh];
#pragma unroll
    for (int a = 0; a < 2; ++a) {
#pragma unroll
      for (int i = 0; i < 8; ++i) {
        int s = s0 + a * 16 + i + 8 * hf;
        float v = acc[a][t][i] + bb;
        if constexpr (VT_OUT)
          out[(((size_t)b * H_ + h) * DH_ + dh) * S_ + s] = (half_t)v;
        else
          out[(((size_t)b * H_ + h) * S_ + s) * DH_ + dh] = (half_t)v;
      }
    }
  }
}

// ---------------- flash attention: one wave per 16-query tile ----------------
// Q,K in [B,H,S,DH]; V transposed in [B,H,DH,S] so P*V B-frags are contiguous.
__global__ __launch_bounds__(256)
void attn_kernel(const half_t* __restrict__ Q, const half_t* __restrict__ K,
                 const half_t* __restrict__ VT, half_t* __restrict__ ctx) {
  __shared__ half_t pbuf[8][16][32];  // per-wave P tile (C-layout f32 -> A-layout f16)
  int wave = threadIdx.x >> 5, lane = threadIdx.x & 31;
  int job = blockIdx.x * 8 + wave;    // b*2048 + h*128 + qt
  int qt = job & 127;
  int h = (job >> 7) & 15;
  int b = job >> 11;
  int s0 = qt * 16;

  const half_t* Qp  = Q  + (((size_t)b * H_ + h) * S_ + s0) * DH_;
  const half_t* K0  = K  + (((size_t)b * H_ + h) * S_) * DH_;
  const half_t* VTh = VT + (((size_t)b * H_ + h) * DH_) * S_;

  v16h aq0 = load_a_frag(Qp, DH_);       // d = 0..31
  v16h aq1 = load_a_frag(Qp + 32, DH_);  // d = 32..63

  float mrun[8], lrun[8];
  v8f o[4];
#pragma unroll
  for (int i = 0; i < 8; ++i) { mrun[i] = -1e30f; lrun[i] = 0.f; }
#pragma unroll
  for (int t = 0; t < 4; ++t) o[t] = v8f_zero();

  const float scale = 0.125f;  // 1/sqrt(64)
  int hf = lane >> 4, nn = lane & 15;

  // double-buffer K score fragments across key tiles
  v16h bk_cur[4];
  bk_cur[0] = load_bT_frag(K0, DH_);
  bk_cur[1] = load_bT_frag(K0 + 32, DH_);
  bk_cur[2] = load_bT_frag(K0 + 16 * DH_, DH_);
  bk_cur[3] = load_bT_frag(K0 + 16 * DH_ + 32, DH_);

  for (int kb = 0; kb < S_; kb += 32) {
    int kn = (kb + 32) & (S_ - 1);

    // V fragments for this tile: issued now, consumed after softmax (latency hidden)
    v16h bv[4];
#pragma unroll
    for (int t = 0; t < 4; ++t)
      bv[t] = load_bT_frag(VTh + (size_t)(t * 16) * S_ + kb, S_);

    // scores: 16x32 tile = Q(16x64) . K(32x64)^T
    v8f c0 = v8f_zero(), c1 = v8f_zero();
    c0 = WMMA_F16(aq0, bk_cur[0], c0);
    c0 = WMMA_F16(aq1, bk_cur[1], c0);
    c1 = WMMA_F16(aq0, bk_cur[2], c1);
    c1 = WMMA_F16(aq1, bk_cur[3], c1);

    // prefetch next tile's K fragments (overlaps softmax VALU below)
    const half_t* Kn = K0 + (size_t)kn * DH_;
    v16h bk_nxt[4];
    bk_nxt[0] = load_bT_frag(Kn, DH_);
    bk_nxt[1] = load_bT_frag(Kn + 32, DH_);
    bk_nxt[2] = load_bT_frag(Kn + 16 * DH_, DH_);
    bk_nxt[3] = load_bT_frag(Kn + 16 * DH_ + 32, DH_);

    // online softmax update
    float p0[8], p1[8];
#pragma unroll
    for (int i = 0; i < 8; ++i) {
      float sa = c0[i] * scale, sb = c1[i] * scale;
      float mx = fmaxf(sa, sb);
#pragma unroll
      for (int off = 1; off < 16; off <<= 1)
        mx = fmaxf(mx, __shfl_xor(mx, off, 32));
      float mnew = fmaxf(mrun[i], mx);
      float corr = __expf(mrun[i] - mnew);
      float e0 = __expf(sa - mnew), e1 = __expf(sb - mnew);
      float rs = e0 + e1;
#pragma unroll
      for (int off = 1; off < 16; off <<= 1)
        rs += __shfl_xor(rs, off, 32);
      lrun[i] = lrun[i] * corr + rs;
      mrun[i] = mnew;
#pragma unroll
      for (int t = 0; t < 4; ++t) o[t][i] = o[t][i] * corr;
      p0[i] = e0; p1[i] = e1;
    }

    // C-layout f32 -> A-layout f16 via per-wave LDS tile (DS in-order per wave)
#pragma unroll
    for (int i = 0; i < 8; ++i) {
      pbuf[wave][i + 8 * hf][nn] = (half_t)p0[i];
      pbuf[wave][i + 8 * hf][16 + nn] = (half_t)p1[i];
    }
    v16h ap = load_a_frag(&pbuf[wave][0][0], 32);

#pragma unroll
    for (int t = 0; t < 4; ++t)
      o[t] = WMMA_F16(ap, bv[t], o[t]);

#pragma unroll
    for (int t = 0; t < 4; ++t) bk_cur[t] = bk_nxt[t];
  }

  // normalize + store ctx [B*S, D] f16
#pragma unroll
  for (int t = 0; t < 4; ++t) {
#pragma unroll
    for (int i = 0; i < 8; ++i) {
      int m = i + 8 * hf;
      float val = o[t][i] / lrun[i];
      ctx[((size_t)(b * S_ + s0 + m)) * D_ + h * DH_ + t * 16 + nn] = (half_t)val;
    }
  }
}

// ---------------- out-proj + residual + LayerNorm (fused) ----------------
// Block: 16 rows x 1024 cols, 512 threads (16 waves); wave w covers cols [w*64, w*64+64).
// The 16-row ctx tile is shared by all waves -> staged to LDS via async copy.
__global__ __launch_bounds__(512)
void oproj_ln_kernel(const half_t* __restrict__ ctx, const half_t* __restrict__ Wo,
                     const float* __restrict__ bo, const float* __restrict__ x,
                     const float* __restrict__ gamma, const float* __restrict__ beta,
                     float* __restrict__ out) {
  __shared__ __align__(16) half_t ctile[16 * D_];  // 32 KB
  __shared__ float ssum[16];
  __shared__ float ssq[16];
  int wave = threadIdx.x >> 5, lane = threadIdx.x & 31;
  int mg0 = blockIdx.x * 16;
  int n0 = wave * 64;

  if (threadIdx.x < 16) { ssum[threadIdx.x] = 0.f; ssq[threadIdx.x] = 0.f; }

  // stage shared ctx tile (16 x 1024 f16 = 32 KB) into LDS
  const char* gsrc = (const char*)(ctx + (size_t)mg0 * D_);
#ifdef HAVE_ASYNC_LDS
  for (int c = threadIdx.x; c < 2048; c += 512)
    __builtin_amdgcn_global_load_async_to_lds_b128(
        (v4i_t*)(gsrc + (size_t)c * 16),
        (v4i_t*)((char*)ctile + c * 16), 0, 0);
  ASYNC_WAIT();
#else
  for (int c = threadIdx.x; c < 2048; c += 512)
    ((float4*)ctile)[c] = ((const float4*)gsrc)[c];
#endif
  __syncthreads();

  v8f acc[4];
#pragma unroll
  for (int t = 0; t < 4; ++t) acc[t] = v8f_zero();

  const half_t* wbase = Wo + (size_t)n0 * D_;

  v16h a_cur, b_cur[4];
  a_cur = load_a_frag(ctile, D_);
#pragma unroll
  for (int t = 0; t < 4; ++t)
    b_cur[t] = load_bT_frag(wbase + (size_t)t * 16 * D_, D_);

  for (int k0 = 0; k0 < D_; k0 += 32) {
    int k1 = (k0 + 32) & (D_ - 1);
    v16h a_nxt, b_nxt[4];
    a_nxt = load_a_frag(ctile + k1, D_);
#pragma unroll
    for (int t = 0; t < 4; ++t)
      b_nxt[t] = load_bT_frag(wbase + (size_t)t * 16 * D_ + k1, D_);

#pragma unroll
    for (int t = 0; t < 4; ++t)
      acc[t] = WMMA_F16(a_cur, b_cur[t], acc[t]);

    a_cur = a_nxt;
#pragma unroll
    for (int t = 0; t < 4; ++t) b_cur[t] = b_nxt[t];
  }

  int hf = lane >> 4, nn = lane & 15;
  // bias + residual; keep values in acc
#pragma unroll
  for (int t = 0; t < 4; ++t) {
    int n = n0 + t * 16 + nn;
#pragma unroll
    for (int i = 0; i < 8; ++i) {
      int m = mg0 + i + 8 * hf;
      acc[t][i] = acc[t][i] + bo[n] + x[(size_t)m * D_ + n];
    }
  }

  // per-row partial stats over this wave's 64 cols, then LDS atomics across 16 waves
#pragma unroll
  for (int i = 0; i < 8; ++i) {
    float s = 0.f, q = 0.f;
#pragma unroll
    for (int t = 0; t < 4; ++t) { float v = acc[t][i]; s += v; q += v * v; }
#pragma unroll
    for (int off = 1; off < 16; off <<= 1) {
      s += __shfl_xor(s, off, 32);
      q += __shfl_xor(q, off, 32);
    }
    if (nn == 0) {
      atomicAdd(&ssum[i + 8 * hf], s);   // ds_add_f32
      atomicAdd(&ssq[i + 8 * hf], q);
    }
  }
  __syncthreads();

#pragma unroll
  for (int i = 0; i < 8; ++i) {
    int mi = i + 8 * hf;
    float mu = ssum[mi] * (1.f / (float)D_);
    float var = ssq[mi] * (1.f / (float)D_) - mu * mu;
    float inv = rsqrtf(var + 1e-5f);
    int m = mg0 + mi;
#pragma unroll
    for (int t = 0; t < 4; ++t) {
      int n = n0 + t * 16 + nn;
      out[(size_t)m * D_ + n] = (acc[t][i] - mu) * inv * gamma[n] + beta[n];
    }
  }
}

extern "C" void kernel_launch(void* const* d_in, const int* in_sizes, int n_in,
                              void* d_out, int out_size, void* d_ws, size_t ws_size,
                              hipStream_t stream) {
  const float* x  = (const float*)d_in[0];
  const float* Wq = (const float*)d_in[1];
  const float* bq = (const float*)d_in[2];
  const float* Wk = (const float*)d_in[3];
  const float* bk = (const float*)d_in[4];
  const float* Wv = (const float*)d_in[5];
  const float* bv = (const float*)d_in[6];
  const float* Wo = (const float*)d_in[7];
  const float* bo = (const float*)d_in[8];
  const float* gamma = (const float*)d_in[9];
  const float* beta  = (const float*)d_in[10];
  float* out = (float*)d_out;

  const size_t nX = (size_t)B_ * S_ * D_;   // 4 Mi elems
  const size_t nW = (size_t)D_ * D_;        // 1 Mi elems

  char* p = (char*)d_ws;
  half_t* xh  = (half_t*)p; p += nX * sizeof(half_t);
  half_t* Wqh = (half_t*)p; p += nW * sizeof(half_t);
  half_t* Wkh = (half_t*)p; p += nW * sizeof(half_t);
  half_t* Wvh = (half_t*)p; p += nW * sizeof(half_t);
  half_t* Woh = (half_t*)p; p += nW * sizeof(half_t);
  half_t* Qh  = (half_t*)p; p += nX * sizeof(half_t);
  half_t* Kh  = (half_t*)p; p += nX * sizeof(half_t);
  half_t* VTh = (half_t*)p; p += nX * sizeof(half_t);
  half_t* ctxh = (half_t*)p; p += nX * sizeof(half_t);
  (void)ws_size; (void)n_in; (void)in_sizes; (void)out_size;

  // fp32 -> fp16 converts
  {
    int n4 = (int)(nX / 4);
    cvt_f32_f16_kernel<<<(n4 + 255) / 256, 256, 0, stream>>>(x, xh, n4);
    int w4 = (int)(nW / 4);
    cvt_f32_f16_kernel<<<(w4 + 255) / 256, 256, 0, stream>>>(Wq, Wqh, w4);
    cvt_f32_f16_kernel<<<(w4 + 255) / 256, 256, 0, stream>>>(Wk, Wkh, w4);
    cvt_f32_f16_kernel<<<(w4 + 255) / 256, 256, 0, stream>>>(Wv, Wvh, w4);
    cvt_f32_f16_kernel<<<(w4 + 255) / 256, 256, 0, stream>>>(Wo, Woh, w4);
  }

  // Q/K/V projections: 256 blocks, 8 waves each (block shares one X tile)
  qkv_proj_kernel<false><<<256, 256, 0, stream>>>(xh, Wqh, bq, Qh);
  qkv_proj_kernel<false><<<256, 256, 0, stream>>>(xh, Wkh, bk, Kh);
  qkv_proj_kernel<true ><<<256, 256, 0, stream>>>(xh, Wvh, bv, VTh);  // V transposed

  // Flash attention: 4096 wave-jobs / 8 waves = 512 blocks
  attn_kernel<<<512, 256, 0, stream>>>(Qh, Kh, VTh, ctxh);

  // Out-proj + residual + LayerNorm: 256 row-tiles, 16 waves each
  oproj_ln_kernel<<<256, 512, 0, stream>>>(ctxh, Woh, bo, x, gamma, beta, out);
}
